// SeqCondAttention_69861938036987
// MI455X (gfx1250) — matrix operand order repaired and checked by reading, not compiled, not run on hardware
//
#include <hip/hip_runtime.h>
#include <hip/hip_bf16.h>

typedef __attribute__((ext_vector_type(16))) _Float16 v16h;
typedef __attribute__((ext_vector_type(8)))  _Float16 v8h;
typedef __attribute__((ext_vector_type(8)))  float    v8f;

// vector/pointer types matching the async-LDS builtin prototype
typedef int v4i_ __attribute__((vector_size(16)));
typedef __attribute__((address_space(1))) v4i_* as1_v4i;
typedef __attribute__((address_space(3))) v4i_* as3_v4i;

// ---------------- problem constants ----------------
#define Bn 4
#define Ln 2048
#define Dn 768
#define Kn 12
#define Hn 64
#define CKn 4
#define DIM_MEM 768           // K*H
#define DIM_MEMK 780          // DIM_MEM + K
#define DSHn 384              // DIM_EXPAND*2
#define DSTn 4608             // K*DSH
#define DIM_FUSED 5388        // DIM_MEM + K + DST
#define ZLD 5504              // DIM_FUSED padded to /128
#define STACKC 1548           // K + 768 + 768
#define SUMC 1536             // 2*K*H
#define YACTC 2304            // K*192
#define BL 8192               // B*L

#if __has_builtin(__builtin_amdgcn_global_load_async_to_lds_b128)
#define HAS_ASYNC_LDS 1
#endif

__device__ __forceinline__ void wait_asynccnt0() {
#if __has_builtin(__builtin_amdgcn_s_wait_asynccnt)
    __builtin_amdgcn_s_wait_asynccnt(0);
#else
    asm volatile("s_wait_asynccnt 0x0" ::: "memory");
#endif
}

__device__ __forceinline__ float siluf(float x) { return x / (1.0f + __expf(-x)); }
__device__ __forceinline__ float sigmf(float x) { return 1.0f / (1.0f + __expf(-x)); }

// ---------------- f32 -> f16 convert with column padding ----------------
__global__ void cvt_f16_pad(const float* __restrict__ src, _Float16* __restrict__ dst,
                            int rows, int cols, int cols_pad) {
    int i = blockIdx.x * blockDim.x + threadIdx.x;
    int total = rows * cols_pad;
    if (i >= total) return;
    int r = i / cols_pad, c = i - r * cols_pad;
    dst[i] = (c < cols) ? (_Float16)src[(size_t)r * cols + c] : (_Float16)0.0f;
}

// ---------------- WMMA GEMM: C[M x Npad](f32) = A[M x Kd](f16) * B[Kd x Npad](f16) ----
// M % 128 == 0, Npad % 128 == 0, Kd % 32 == 0. Row-major A,B,C. ldc == Npad.
// 256 threads = 8 waves; block tile 128x128; wave tile 32x64 (8 WMMAs / K-step).
// Double-buffered LDS; A staged via GLOBAL_LOAD_ASYNC_TO_LDS (ASYNCcnt) when available.
#define ASTR 40   // LDS row stride (halves): 80B -> 16B aligned, conflict-free
#define BSTR 40   // LDS column stride (halves)
__global__ __launch_bounds__(256) void wmma_gemm_f16f32(
    const _Float16* __restrict__ A, const _Float16* __restrict__ B,
    float* __restrict__ C, int M, int Npad, int Kd) {
    __shared__ alignas(16) _Float16 As[2][128 * ASTR];   // row-major 128x32 (+pad)
    __shared__ alignas(16) _Float16 Bs[2][128 * BSTR];   // column-major: Bs[col*BSTR + k]

    const int tid  = threadIdx.x;
    const int wave = tid >> 5;
    const int lane = tid & 31;
    const int m16  = lane & 15;
    const int hi   = lane >> 4;
    const int wm   = wave & 3;     // M strip: rows wm*32 .. +31
    const int wn   = wave >> 2;    // N half:  cols wn*64 .. +63
    const int bm   = blockIdx.y * 128;
    const int bn   = blockIdx.x * 128;

    v8f acc[2][4] = {};

    // ---- staging helpers -------------------------------------------------
#ifdef HAS_ASYNC_LDS
    auto issueA = [&](int buf, int k0) {   // 128x32 halves, 2x16B async copies/thread
#pragma unroll
        for (int i = 0; i < 2; ++i) {
            int ch = tid * 2 + i;          // 0..511
            int r  = ch >> 2;
            int c8 = (ch & 3) << 3;
            _Float16* gsrc = (_Float16*)&A[(size_t)(bm + r) * Kd + k0 + c8];
            _Float16* ldst = &As[buf][r * ASTR + c8];
            __builtin_amdgcn_global_load_async_to_lds_b128(
                (as1_v4i)gsrc, (as3_v4i)ldst, 0, 0);
        }
    };
#else
    float4 av[2];
    auto loadA_regs = [&](int k0) {
#pragma unroll
        for (int i = 0; i < 2; ++i) {
            int ch = tid * 2 + i;
            int r  = ch >> 2;
            int c8 = (ch & 3) << 3;
            av[i] = *(const float4*)(&A[(size_t)(bm + r) * Kd + k0 + c8]);
        }
    };
    auto storeA_lds = [&](int buf) {
#pragma unroll
        for (int i = 0; i < 2; ++i) {
            int ch = tid * 2 + i;
            int r  = ch >> 2;
            int c8 = (ch & 3) << 3;
            *(float4*)(&As[buf][r * ASTR + c8]) = av[i];
        }
    };
#endif
    float4 bv[2];
    auto loadB_regs = [&](int k0) {        // 32x128 halves
#pragma unroll
        for (int i = 0; i < 2; ++i) {
            int ch = tid * 2 + i;
            int r  = ch >> 4;              // k within tile
            int c8 = (ch & 15) << 3;       // col base
            bv[i] = *(const float4*)(&B[(size_t)(k0 + r) * Npad + bn + c8]);
        }
    };
    auto storeB_lds = [&](int buf) {       // transpose into column-major LDS
#pragma unroll
        for (int i = 0; i < 2; ++i) {
            int ch = tid * 2 + i;
            int r  = ch >> 4;
            int c8 = (ch & 15) << 3;
            const _Float16* hv = (const _Float16*)&bv[i];
#pragma unroll
            for (int j = 0; j < 8; ++j) Bs[buf][(c8 + j) * BSTR + r] = hv[j];
        }
    };
    auto compute = [&](int buf) {
#pragma unroll
        for (int f = 0; f < 2; ++f) {
            // A fragment (16x32): halves i -> K = 16*hi + i
            const _Float16* arow = &As[buf][(wm * 32 + f * 16 + m16) * ASTR + hi * 16];
            v8h alo = *(const v8h*)(arow);
            v8h ahi = *(const v8h*)(arow + 8);
            v16h afrag = __builtin_shufflevector(alo, ahi,
                0,1,2,3,4,5,6,7,8,9,10,11,12,13,14,15);
#pragma unroll
            for (int t = 0; t < 4; ++t) {
                // B fragment (32x16): column = lane&15, halves i -> K = 16*hi + i
                const _Float16* bcol = &Bs[buf][(wn * 64 + t * 16 + m16) * BSTR + hi * 16];
                v8h blo = *(const v8h*)(bcol);
                v8h bhi = *(const v8h*)(bcol + 8);
                v16h bfrag = __builtin_shufflevector(blo, bhi,
                    0,1,2,3,4,5,6,7,8,9,10,11,12,13,14,15);
                acc[f][t] = __builtin_amdgcn_wmma_f32_16x16x32_f16(
                    false, afrag, false, bfrag, (short)0, acc[f][t], false, false);
            }
        }
    };

    // ---- prologue: stage tile 0 into buffer 0 ---------------------------
#ifdef HAS_ASYNC_LDS
    issueA(0, 0);
#else
    loadA_regs(0);
    storeA_lds(0);
#endif
    loadB_regs(0);
    storeB_lds(0);
#ifdef HAS_ASYNC_LDS
    wait_asynccnt0();
#endif
    __syncthreads();

    // ---- main pipeline: compute buf `cur`, stage buf `cur^1` ------------
    int cur = 0;
    for (int k0 = 0; k0 < Kd; k0 += 32) {
        const int nxt = cur ^ 1;
        const bool more = (k0 + 32) < Kd;
        if (more) {
#ifdef HAS_ASYNC_LDS
            issueA(nxt, k0 + 32);          // async global -> LDS, overlaps WMMAs
#else
            loadA_regs(k0 + 32);
#endif
            loadB_regs(k0 + 32);
        }
        if (k0 + 64 < Kd)                  // keep a far prefetch in flight
            __builtin_prefetch(&A[(size_t)(bm + (tid >> 1)) * Kd + k0 + 64], 0, 3);

        compute(cur);

        if (more) {
            storeB_lds(nxt);
#ifndef HAS_ASYNC_LDS
            storeA_lds(nxt);
#endif
#ifdef HAS_ASYNC_LDS
            wait_asynccnt0();
#endif
        }
        __syncthreads();
        cur = nxt;
    }

    // ---- C store: VGPR r -> row = r + 8*hi ; col = lane&15 --------------
    const int rowb = bm + wm * 32 + (hi << 3);
    const int n0   = bn + wn * 64 + m16;
#pragma unroll
    for (int f = 0; f < 2; ++f)
#pragma unroll
        for (int t = 0; t < 4; ++t)
#pragma unroll
            for (int r = 0; r < 8; ++r)
                C[(size_t)(rowb + f * 16 + r) * Npad + n0 + t * 16] = acc[f][t][r];
}

// ---------------- stage: conv(CK=4)+SiLU+RMSnorm+p_w+re/im -> stack -------------
// one 64-thread block per (b,l,k); h = threadIdx.x
__global__ __launch_bounds__(64) void stage_kernel(
    const float* __restrict__ zf, const float* __restrict__ conv_w,
    const float* __restrict__ rms_scale, const float* __restrict__ theta_raw,
    const float* __restrict__ decay_slopes, const float* __restrict__ score_scale,
    const float* __restrict__ tanh_scale, float* __restrict__ stack) {
    __shared__ float red[64];
    __shared__ float s_sh;
    int bid = blockIdx.x;
    int k = bid % Kn;
    int l = (bid / Kn) % Ln;
    int b = bid / (Kn * Ln);
    int h = threadIdx.x;
    int c = k * Hn + h;

    float acc = 0.0f;
#pragma unroll
    for (int j = 0; j < CKn; ++j) {
        int lp = l - (CKn - 1) + j;
        if (lp >= 0)
            acc += zf[((size_t)(b * Ln + lp)) * ZLD + c] * conv_w[j * DIM_MEMK + c];
    }
    float kv = siluf(acc);

    if (h == 0) {
        int cs = DIM_MEM + k;
        float sacc = 0.0f;
#pragma unroll
        for (int j = 0; j < CKn; ++j) {
            int lp = l - (CKn - 1) + j;
            if (lp >= 0)
                sacc += zf[((size_t)(b * Ln + lp)) * ZLD + cs] * conv_w[j * DIM_MEMK + cs];
        }
        s_sh = siluf(sacc);
    }
    red[h] = kv * kv;
    __syncthreads();
    for (int off = 32; off > 0; off >>= 1) {
        if (h < off) red[h] += red[h + off];
        __syncthreads();
    }
    float kval = kv * rsqrtf(red[0] * (1.0f / 64.0f) + 1e-6f) * rms_scale[h];

    float theta = 0.001f + 2.999f * sigmf(theta_raw[k * Hn + h]);
    float slope = log1pf(__expf(decay_slopes[k]));
    float ltw   = -slope * (float)(Ln - 1 - l);
    float lp_   = fminf(fmaxf(score_scale[k] * s_sh, -10.0f), 10.0f);
    float p_w   = __expf(lp_ + ltw);

    float phi = tanhf(kval * tanh_scale[k]) * theta;
    float kvw = kval * p_w;
    float re  = kvw * __cosf(phi);
    float im  = kvw * __sinf(phi);

    size_t base = ((size_t)(b * Ln + l)) * STACKC;
    stack[base + Kn + k * Hn + h] = re;
    stack[base + Kn + DIM_MEM + k * Hn + h] = im;
    if (h == 0) stack[base + k] = p_w;
}

// ---------------- in-place cumsum over L, one thread per (b, channel) -----------
__global__ void cumsum_kernel(float* __restrict__ stack) {
    int idx = blockIdx.x * blockDim.x + threadIdx.x;
    if (idx >= Bn * STACKC) return;
    int b = idx / STACKC, c = idx - b * STACKC;
    float run = 0.0f;
    size_t p = ((size_t)b * Ln) * STACKC + c;
    for (int l = 0; l < Ln; ++l, p += STACKC) {
        run += stack[p];
        stack[p] = run;
    }
}

// ---------------- summary_flat (f16) ----------------
__global__ void summarize_kernel(const float* __restrict__ cs, _Float16* __restrict__ summ) {
    int i = blockIdx.x * blockDim.x + threadIdx.x;
    if (i >= BL * SUMC) return;
    int bl = i / SUMC, s = i - bl * SUMC;
    int kh = s >> 1, ri = s & 1, k = kh >> 6;
    size_t base = (size_t)bl * STACKC;
    float inv = 1.0f / fmaxf(cs[base + k], 1e-4f);
    float v = cs[base + Kn + ri * DIM_MEM + kh] * inv;
    summ[i] = (_Float16)v;
}

// ---------------- gated epilogue -> y_act (f16) ----------------
__global__ void epilogue_kernel(const float* __restrict__ ysp, const float* __restrict__ zf,
                                _Float16* __restrict__ yact) {
    int i = blockIdx.x * blockDim.x + threadIdx.x;
    if (i >= BL * YACTC) return;
    int bl = i / YACTC, s = i - bl * YACTC;
    int k = s / 192, j = s - k * 192;
    size_t sb = (size_t)bl * DSTn + k * DSHn;
    size_t zb = (size_t)bl * ZLD + DIM_MEMK + k * DSHn;
    float val = ysp[sb + j] + zf[zb + j];
    float gat = ysp[sb + 192 + j] + zf[zb + 192 + j];
    yact[i] = (_Float16)(val * siluf(gat));
}

// ---------------- launch ----------------
extern "C" void kernel_launch(void* const* d_in, const int* in_sizes, int n_in,
                              void* d_out, int out_size, void* d_ws, size_t ws_size,
                              hipStream_t stream) {
    const float* x      = (const float*)d_in[0];
    const float* W_in   = (const float*)d_in[1];
    const float* conv_w = (const float*)d_in[2];
    const float* rms_s  = (const float*)d_in[3];
    const float* th_raw = (const float*)d_in[4];
    const float* dslope = (const float*)d_in[5];
    const float* sc_s   = (const float*)d_in[6];
    const float* tn_s   = (const float*)d_in[7];
    const float* W_sw   = (const float*)d_in[8];
    const float* W_out  = (const float*)d_in[9];
    float* out = (float*)d_out;

    char* p = (char*)d_ws;
    auto carve = [&](size_t bytes) {
        void* q = (void*)p;
        p += (bytes + 255) & ~(size_t)255;
        return q;
    };
    _Float16* xh   = (_Float16*)carve((size_t)BL * Dn * 2);
    _Float16* Wih  = (_Float16*)carve((size_t)Dn * ZLD * 2);
    _Float16* Wsh  = (_Float16*)carve((size_t)SUMC * DSTn * 2);
    _Float16* Woh  = (_Float16*)carve((size_t)YACTC * Dn * 2);
    float*    zf   = (float*)   carve((size_t)BL * ZLD * 4);
    float*    st   = (float*)   carve((size_t)BL * STACKC * 4);
    _Float16* summ = (_Float16*)carve((size_t)BL * SUMC * 2);
    float*    ysp  = (float*)   carve((size_t)BL * DSTn * 4);
    _Float16* yact = (_Float16*)carve((size_t)BL * YACTC * 2);

    auto nblk = [](size_t n, int t) { return (unsigned)((n + t - 1) / t); };

    // f32 -> f16 conversions (W_in zero-padded 5388 -> 5504 cols)
    cvt_f16_pad<<<nblk((size_t)BL * Dn, 256), 256, 0, stream>>>(x, xh, BL, Dn, Dn);
    cvt_f16_pad<<<nblk((size_t)Dn * ZLD, 256), 256, 0, stream>>>(W_in, Wih, Dn, DIM_FUSED, ZLD);
    cvt_f16_pad<<<nblk((size_t)SUMC * DSTn, 256), 256, 0, stream>>>(W_sw, Wsh, SUMC, DSTn, DSTn);
    cvt_f16_pad<<<nblk((size_t)YACTC * Dn, 256), 256, 0, stream>>>(W_out, Woh, YACTC, Dn, Dn);

    // GEMM 1: z_fused = x @ W_in  (8192 x 768 x 5504)
    {
        dim3 g(ZLD / 128, BL / 128);
        wmma_gemm_f16f32<<<g, 256, 0, stream>>>(xh, Wih, zf, BL, ZLD, Dn);
    }
    // conv + silu + rms + spectral features
    stage_kernel<<<Bn * Ln * Kn, 64, 0, stream>>>(zf, conv_w, rms_s, th_raw, dslope,
                                                  sc_s, tn_s, st);
    // cumsum over L (in place)
    cumsum_kernel<<<nblk(Bn * STACKC, 256), 256, 0, stream>>>(st);
    // normalized summaries -> f16
    summarize_kernel<<<nblk((size_t)BL * SUMC, 256), 256, 0, stream>>>(st, summ);
    // GEMM 2: y_spectral = summary @ W_sw  (8192 x 1536 x 4608)
    {
        dim3 g(DSTn / 128, BL / 128);
        wmma_gemm_f16f32<<<g, 256, 0, stream>>>(summ, Wsh, ysp, BL, DSTn, SUMC);
    }
    // gated combine with skip path
    epilogue_kernel<<<nblk((size_t)BL * YACTC, 256), 256, 0, stream>>>(ysp, zf, yact);
    // GEMM 3: out = y_act @ W_out  (8192 x 2304 x 768), straight into d_out
    {
        dim3 g(Dn / 128, BL / 128);
        wmma_gemm_f16f32<<<g, 256, 0, stream>>>(yact, Woh, out, BL, Dn, YACTC);
    }
}
